// MeanAggregator_11355893531250
// MI455X (gfx1250) — matrix operand order repaired and checked by reading, not compiled, not run on hardware
//
#include <hip/hip_runtime.h>
#include <hip/hip_bf16.h>

#define N_NODES   4096
#define NEIGH     25
#define F_DIM     128
#define H_HEADS   8
#define D_DIM     16

typedef __attribute__((ext_vector_type(16))) _Float16 v16h;
typedef __attribute__((ext_vector_type(8)))  float    v8f;

// --- CDNA5 async global->LDS path (probe via __has_builtin, safe fallback) --
#if defined(__has_builtin)
#if __has_builtin(__builtin_amdgcn_global_load_async_to_lds_b32)
#define USE_ASYNC_LDS 1
#endif
#endif

#if defined(USE_ASYNC_LDS)
#if __has_builtin(__builtin_amdgcn_s_wait_asynccnt)
#define WAIT_ASYNC(n) __builtin_amdgcn_s_wait_asynccnt(n)
#else
#define WAIT_ASYNC(n) asm volatile("s_wait_asynccnt %0" :: "i"(n) : "memory")
#endif
// builtin signature (per hipcc diagnostic): AS1 int* global src, AS3 int* LDS
// dst, imm offset, imm cpol
#define ASYNC_COPY_B32(gptr, lptr)                                            \
    __builtin_amdgcn_global_load_async_to_lds_b32(                            \
        (__attribute__((address_space(1))) int*)(gptr),                       \
        (__attribute__((address_space(3))) int*)(lptr), 0, 0)
#else
#define WAIT_ASYNC(n) do {} while (0)
#define ASYNC_COPY_B32(gptr, lptr) do { *(lptr) = *(gptr); } while (0)
#endif

// ---------------------------------------------------------------------------
// K0: gather node features + mean of neighbor features
// ---------------------------------------------------------------------------
__global__ void gather_mean_kernel(const float* __restrict__ features,
                                   const int* __restrict__ node,
                                   const int* __restrict__ neighbor,
                                   float* __restrict__ node_feat,
                                   float* __restrict__ neigh_feat) {
    int i = blockIdx.x;
    int f = threadIdx.x;
    node_feat[i * F_DIM + f] = features[node[i] * F_DIM + f];
    float acc = 0.0f;
    #pragma unroll 5
    for (int t = 0; t < NEIGH; ++t) {
        acc += features[neighbor[i * NEIGH + t] * F_DIM + f];
    }
    neigh_feat[i * F_DIM + f] = acc * (1.0f / (float)NEIGH);
}

// ---------------------------------------------------------------------------
// K1: projection GEMM: out = X (4096x128) @ W (128x128), via WMMA f16->f32
// grid: (256) i-tiles of 16 rows; block: 256 = 8 waves, wave w owns cols
// [16w, 16w+16) (== head w). Also writes f16 copy (for the apply pass).
// A-operand (16x32 f16): lane row m=lane&15, elem e -> k = e+8*(e>>3)+8*(lane>>4)
// B-operand (32x16 f16): lane col n=lane&15, elem e -> k = e+8*(e>>3)+8*(lane>>4)
// C/D (16x16 f32): lane col n=lane&15, VGPR r -> row r+8*(lane>>4)
// ---------------------------------------------------------------------------
__global__ void gemm_proj_kernel(const float* __restrict__ X,
                                 const float* __restrict__ W,
                                 float* __restrict__ out32,
                                 _Float16* __restrict__ out16) {
    int lane = threadIdx.x & 31;
    int wave = threadIdx.x >> 5;
    int i0   = blockIdx.x * 16;
    int m    = lane & 15;
    int hi   = lane >> 4;
    int col  = wave * 16 + m;

    v8f c = {};
    #pragma unroll
    for (int kc = 0; kc < 4; ++kc) {
        v16h a, b;
        #pragma unroll
        for (int e = 0; e < 16; ++e) {
            int ka = e + ((e >> 3) << 3) + (hi << 3);
            a[e] = (_Float16)X[(i0 + m) * F_DIM + kc * 32 + ka];
            int kb = e + ((e >> 3) << 3) + (hi << 3);
            b[e] = (_Float16)W[(kc * 32 + kb) * F_DIM + col];
        }
        c = __builtin_amdgcn_wmma_f32_16x16x32_f16(
                false, a, false, b, (short)0, c, false, false);
    }
    #pragma unroll
    for (int r = 0; r < 8; ++r) {
        int mr = r + (hi << 3);
        float v = c[r];
        out32[(i0 + mr) * F_DIM + col] = v;
        if (out16) out16[(i0 + mr) * F_DIM + col] = (_Float16)v;
    }
}

// ---------------------------------------------------------------------------
// K2: per-head attention logits
// ---------------------------------------------------------------------------
__global__ void att_logits_kernel(const float* __restrict__ nf,
                                  const float* __restrict__ gf,
                                  const float* __restrict__ aw_s,
                                  const float* __restrict__ aw_n,
                                  float* __restrict__ att_s,
                                  float* __restrict__ att_n) {
    int idx = blockIdx.x * blockDim.x + threadIdx.x;
    int h = idx >> 12;
    int i = idx & (N_NODES - 1);
    float s = 0.0f, n = 0.0f;
    #pragma unroll
    for (int d = 0; d < D_DIM; ++d) {
        s += nf[i * F_DIM + h * D_DIM + d] * aw_s[h * D_DIM + d];
        n += gf[i * F_DIM + h * D_DIM + d] * aw_n[h * D_DIM + d];
    }
    att_s[h * N_NODES + i] = s;
    att_n[h * N_NODES + i] = n;
}

// ---------------------------------------------------------------------------
// K3: streaming-softmax pass 1: per (head, row) max + denominator.
// Masked entries contribute exactly 0 (exp underflow) -> skip non-edges.
// ---------------------------------------------------------------------------
__global__ void row_stats_kernel(const float* __restrict__ A,
                                 const float* __restrict__ att_s,
                                 const float* __restrict__ att_n,
                                 float* __restrict__ mrow,
                                 float* __restrict__ lrow) {
    __shared__ float rm[256];
    __shared__ float rs[256];
    int i   = blockIdx.x;
    int tid = threadIdx.x;

    float si[H_HEADS], m[H_HEADS], s[H_HEADS];
    #pragma unroll
    for (int h = 0; h < H_HEADS; ++h) {
        si[h] = att_s[h * N_NODES + i];
        m[h]  = -3.0e38f;
        s[h]  = 0.0f;
    }
    for (int j = tid; j < N_NODES; j += 256) {
        __builtin_prefetch(&A[i * N_NODES + j + 2048], 0, 1);
        float a = A[i * N_NODES + j];
        if (a != 0.0f) {
            #pragma unroll
            for (int h = 0; h < H_HEADS; ++h) {
                float x = si[h] + att_n[h * N_NODES + j];
                x = (x > 0.0f) ? x : 0.2f * x;           // leaky_relu(0.2)
                float mo = m[h];
                if (x > mo) { s[h] = s[h] * __expf(mo - x) + 1.0f; m[h] = x; }
                else        { s[h] += __expf(x - mo); }
            }
        }
    }
    for (int h = 0; h < H_HEADS; ++h) {
        rm[tid] = m[h];
        rs[tid] = s[h];
        __syncthreads();
        for (int off = 128; off > 0; off >>= 1) {
            if (tid < off) {
                float m1 = rm[tid], m2 = rm[tid + off];
                float s1 = rs[tid], s2 = rs[tid + off];
                float mm = fmaxf(m1, m2);
                rm[tid] = mm;
                rs[tid] = s1 * __expf(m1 - mm) + s2 * __expf(m2 - mm);
            }
            __syncthreads();
        }
        if (tid == 0) {
            mrow[h * N_NODES + i] = rm[0];
            lrow[h * N_NODES + i] = rs[0];
        }
        __syncthreads();
    }
}

// ---------------------------------------------------------------------------
// K4: pass 2, fused normalize + apply: out[i,h,:] = relu(P[h,i,:] @ nf[:,h,:])
// grid: (256) i-tiles of 16 rows; block: 256 = 8 waves, wave = head.
// A tile (16x32 f32) + att_n slice (8x32) double-buffered in LDS; the A tile
// is staged with GLOBAL_LOAD_ASYNC_TO_LDS_B32 (ASYNCcnt pipeline) when the
// toolchain exposes it, overlapping HBM latency with the WMMA accumulation.
// ---------------------------------------------------------------------------
__global__ void attn_apply_kernel(const float* __restrict__ A,
                                  const float* __restrict__ att_s,
                                  const float* __restrict__ att_n,
                                  const float* __restrict__ mrow,
                                  const float* __restrict__ lrow,
                                  const _Float16* __restrict__ nf16,
                                  float* __restrict__ out) {
    __shared__ float At[2][16][33];       // +1 pad: avoid bank conflicts
    __shared__ float An[2][H_HEADS][32];
    int tid  = threadIdx.x;
    int lane = tid & 31;
    int h    = tid >> 5;                  // wave index == head
    int i0   = blockIdx.x * 16;
    int m    = lane & 15;
    int hi   = lane >> 4;

    float s_i   = att_s[h * N_NODES + i0 + m];
    float mv    = mrow [h * N_NODES + i0 + m];
    float inv_l = 1.0f / lrow[h * N_NODES + i0 + m];

    // stage A tile (async when available) + att_n slice (sync) for `tile`
    auto stage = [&](int tile, int buf) {
        int r0 = tid >> 5, c0 = tid & 31;
        ASYNC_COPY_B32(&A[(i0 + r0) * N_NODES + tile * 32 + c0],
                       &At[buf][r0][c0]);
        int idx2 = tid + 256;
        int r1 = idx2 >> 5, c1 = idx2 & 31;
        ASYNC_COPY_B32(&A[(i0 + r1) * N_NODES + tile * 32 + c1],
                       &At[buf][r1][c1]);
        // 8 heads x 32 cols == 256 threads: one element each
        An[buf][r0][c0] = att_n[r0 * N_NODES + tile * 32 + c0];
    };

    const int NT = N_NODES / 32;          // 128 K-chunks
    stage(0, 0);

    v8f c = {};
    for (int t = 0; t < NT; ++t) {
        int cur = t & 1;
        if (t + 1 < NT) {
            stage(t + 1, cur ^ 1);        // prefetch next tile into other buf
            WAIT_ASYNC(2);                // oldest 2 (current tile) landed
        } else {
            WAIT_ASYNC(0);
        }
        __syncthreads();                  // all waves' current tile visible

        v16h pa, bm;
        #pragma unroll
        for (int e = 0; e < 16; ++e) {
            int ka = e + ((e >> 3) << 3) + (hi << 3);   // A-operand k index
            float a = At[cur][m][ka];
            float p = 0.0f;
            if (a != 0.0f) {
                float x = s_i + An[cur][h][ka];
                x = (x > 0.0f) ? x : 0.2f * x;
                p = __expf(x - mv) * inv_l;             // softmax prob
            }
            pa[e] = (_Float16)p;
            int kb = e + ((e >> 3) << 3) + (hi << 3);   // B-operand k index
            bm[e] = nf16[(t * 32 + kb) * F_DIM + h * D_DIM + m];
        }
        c = __builtin_amdgcn_wmma_f32_16x16x32_f16(
                false, pa, false, bm, (short)0, c, false, false);

        __syncthreads();                  // reads done before buf is rewritten
    }
    #pragma unroll
    for (int r = 0; r < 8; ++r) {
        int mr = r + (hi << 3);
        out[(i0 + mr) * F_DIM + h * D_DIM + m] = fmaxf(c[r], 0.0f);
    }
}

// ---------------------------------------------------------------------------
extern "C" void kernel_launch(void* const* d_in, const int* in_sizes, int n_in,
                              void* d_out, int out_size, void* d_ws, size_t ws_size,
                              hipStream_t stream) {
    const float* A        = (const float*)d_in[0];   // 4096*4096
    const float* features = (const float*)d_in[1];   // 100000*128
    const int*   node     = (const int*)  d_in[2];   // 4096
    const int*   neighbor = (const int*)  d_in[3];   // 4096*25
    const float* W        = (const float*)d_in[4];   // 128*128
    const float* aw_s     = (const float*)d_in[5];   // 8*16
    const float* aw_n     = (const float*)d_in[6];   // 8*16
    float*       out      = (float*)d_out;           // 4096*128

    float* ws         = (float*)d_ws;
    float* node_feat  = ws;                               // 4096*128
    float* neigh_feat = node_feat  + N_NODES * F_DIM;     // 4096*128
    float* nf         = neigh_feat + N_NODES * F_DIM;     // 4096*128
    float* gf         = nf         + N_NODES * F_DIM;     // 4096*128
    float* att_s      = gf         + N_NODES * F_DIM;     // 8*4096
    float* att_n      = att_s      + H_HEADS * N_NODES;   // 8*4096
    float* mrow       = att_n      + H_HEADS * N_NODES;   // 8*4096
    float* lrow       = mrow       + H_HEADS * N_NODES;   // 8*4096
    _Float16* nf16    = (_Float16*)(lrow + H_HEADS * N_NODES); // 4096*128

    gather_mean_kernel<<<N_NODES, F_DIM, 0, stream>>>(
        features, node, neighbor, node_feat, neigh_feat);

    gemm_proj_kernel<<<N_NODES / 16, 256, 0, stream>>>(node_feat, W, nf, nf16);
    gemm_proj_kernel<<<N_NODES / 16, 256, 0, stream>>>(neigh_feat, W, gf, (_Float16*)nullptr);

    att_logits_kernel<<<(H_HEADS * N_NODES) / 256, 256, 0, stream>>>(
        nf, gf, aw_s, aw_n, att_s, att_n);

    row_stats_kernel<<<N_NODES, 256, 0, stream>>>(A, att_s, att_n, mrow, lrow);

    attn_apply_kernel<<<N_NODES / 16, 256, 0, stream>>>(
        A, att_s, att_n, mrow, lrow, nf16, out);
}
